// MLPDecoder_3959959847148
// MI455X (gfx1250) — matrix operand order repaired and checked by reading, not compiled, not run on hardware
//
#include <hip/hip_runtime.h>

// ---------------- types ----------------
typedef __attribute__((ext_vector_type(16))) __bf16        v16bf;
typedef __attribute__((ext_vector_type(8)))  float         v8f;
typedef __attribute__((ext_vector_type(4)))  unsigned int  u32x4;
typedef __attribute__((ext_vector_type(4)))  float         f32x4;
typedef __attribute__((ext_vector_type(4)))  unsigned short u16x4;

union Frag { v16bf v; u32x4 u[2]; };   // 32 bytes: one 16x32 bf16 A/B fragment

__device__ __forceinline__ unsigned short f2bf(float x) {
    union { float f; unsigned int u; } cv; cv.f = x;
    unsigned int u = cv.u;
    unsigned int r = u + 0x7FFFu + ((u >> 16) & 1u);   // round to nearest even
    return (unsigned short)(r >> 16);
}

__device__ __forceinline__ void lds_fence() {
    asm volatile("s_wait_dscnt 0" ::: "memory");
}

// ---------------- prep: f32 -> bf16 (vectorized x4) ----------------
__global__ void cvt_bf16_kernel(const float* __restrict__ in,
                                unsigned short* __restrict__ out, int n4) {
    int i = blockIdx.x * blockDim.x + threadIdx.x;
    if (i < n4) {
        f32x4 v = ((const f32x4*)in)[i];
        u16x4 o;
        o.x = f2bf(v.x); o.y = f2bf(v.y); o.z = f2bf(v.z); o.w = f2bf(v.w);
        ((u16x4*)out)[i] = o;
    }
}

// ---------------- prep: W[K][N] f32 -> WT[N][K] bf16 ----------------
__global__ void transpose_bf16_kernel(const float* __restrict__ in,
                                      unsigned short* __restrict__ out,
                                      int K, int N) {
    int idx = blockIdx.x * blockDim.x + threadIdx.x;
    if (idx < K * N) {
        int n = idx / K;
        int k = idx - n * K;
        out[idx] = f2bf(in[k * N + n]);
    }
}

// ---------------- main fused edge-MLP kernel ----------------
// Block: 256 threads = 8 waves, 128 edges (16 edges per wave).
// LDS: [0,128K) W1T bf16, [128K,144K) W2T bf16, [144K,176K) 4KB/wave stage.
#define SM_W1   0
#define SM_W2   131072
#define SM_STG  147456
#define SM_TOT  180224

__device__ __forceinline__ u32x4 ld_feat_chunk(const unsigned short* __restrict__ drugp,
                                               const unsigned short* __restrict__ disp,
                                               int k) {
    const unsigned short* p = (k < 256) ? (drugp + k) : (disp + (k - 256));
    return *(const u32x4*)p;   // 8 contiguous bf16 = b128
}

__global__ void __launch_bounds__(256, 1)
mlp_edge_kernel(const unsigned short* __restrict__ drug_bf,
                const unsigned short* __restrict__ dis_bf,
                const int* __restrict__ src_idx,
                const int* __restrict__ dst_idx,
                const unsigned short* __restrict__ w1t,   // [128][512] bf16
                const unsigned short* __restrict__ w2t,   // [64][128]  bf16
                const float* __restrict__ b1,
                const float* __restrict__ b2,
                const float* __restrict__ w3,             // [64] f32
                const float* __restrict__ b3,
                float* __restrict__ out, int E) {
    extern __shared__ char smem[];
    unsigned short* sW1 = (unsigned short*)(smem + SM_W1);
    unsigned short* sW2 = (unsigned short*)(smem + SM_W2);

    const int tid  = threadIdx.x;
    const int wv   = tid >> 5;
    const int lane = tid & 31;
    const int row  = lane & 15;     // M within 16x16 tile
    const int kg   = lane >> 4;     // K half-group selector

    unsigned short* sH1 = (unsigned short*)(smem + SM_STG + wv * 4096); // 16x128 bf16
    float*          sH2 = (float*)        (smem + SM_STG + wv * 4096); // 16x64  f32

    // ---- stage W1T (8192 x b128) and W2T (1024 x b128) into LDS ----
    {
        const u32x4* g1 = (const u32x4*)w1t;
        u32x4*       s1 = (u32x4*)sW1;
        #pragma unroll 4
        for (int i = tid; i < 8192; i += 256) s1[i] = g1[i];
        const u32x4* g2 = (const u32x4*)w2t;
        u32x4*       s2 = (u32x4*)sW2;
        #pragma unroll
        for (int i = tid; i < 1024; i += 256) s2[i] = g2[i];
    }
    __syncthreads();

    // ---- gather A fragments: 16 edges x 512 bf16 (A layout, ISA 7.12.2) ----
    const int e0 = blockIdx.x * 128 + wv * 16;
    int e = e0 + row; if (e > E - 1) e = E - 1;
    const unsigned short* drugp = drug_bf + (long)src_idx[e] * 256;
    const unsigned short* disp  = dis_bf  + (long)dst_idx[e] * 256;

    Frag a[16];
    #pragma unroll
    for (int kf = 0; kf < 16; ++kf) {
        int c0 = kf * 32 + 8 * kg;
        a[kf].u[0] = ld_feat_chunk(drugp, disp, c0);
        a[kf].u[1] = ld_feat_chunk(drugp, disp, c0 + 16);
    }

    // ---- layer 1: [16x512] x [512x128] + b1, ReLU, stage bf16 ----
    #pragma unroll
    for (int nt = 0; nt < 8; ++nt) {
        const int n = nt * 16 + row;
        const float bv = b1[n];
        v8f acc = {bv, bv, bv, bv, bv, bv, bv, bv};
        #pragma unroll
        for (int kf = 0; kf < 16; ++kf) {
            Frag bfr;
            const u32x4* p = (const u32x4*)(sW1 + n * 512 + kf * 32 + kg * 8);
            bfr.u[0] = p[0]; bfr.u[1] = p[2];
            acc = __builtin_amdgcn_wmma_f32_16x16x32_bf16(
                false, a[kf].v, false, bfr.v, (short)0, acc, false, false);
        }
        #pragma unroll
        for (int i = 0; i < 8; ++i) {               // C layout: m = 8*kg + i
            sH1[(kg * 8 + i) * 128 + n] = f2bf(fmaxf(acc[i], 0.0f));
        }
    }
    lds_fence();

    // ---- reload H1 strip as layer-2 A fragments ----
    Frag a2[4];
    #pragma unroll
    for (int kf = 0; kf < 4; ++kf) {
        int c0 = kf * 32 + 8 * kg;
        a2[kf].u[0] = *(const u32x4*)(sH1 + row * 128 + c0);
        a2[kf].u[1] = *(const u32x4*)(sH1 + row * 128 + c0 + 16);
    }
    lds_fence();

    // ---- layer 2: [16x128] x [128x64] + b2, ReLU, stage f32 ----
    #pragma unroll
    for (int nt = 0; nt < 4; ++nt) {
        const int n = nt * 16 + row;
        const float bv = b2[n];
        v8f acc = {bv, bv, bv, bv, bv, bv, bv, bv};
        #pragma unroll
        for (int kf = 0; kf < 4; ++kf) {
            Frag bfr;
            const u32x4* p = (const u32x4*)(sW2 + n * 128 + kf * 32 + kg * 8);
            bfr.u[0] = p[0]; bfr.u[1] = p[2];
            acc = __builtin_amdgcn_wmma_f32_16x16x32_bf16(
                false, a2[kf].v, false, bfr.v, (short)0, acc, false, false);
        }
        #pragma unroll
        for (int i = 0; i < 8; ++i) {
            sH2[(kg * 8 + i) * 64 + n] = fmaxf(acc[i], 0.0f);
        }
    }
    lds_fence();

    // ---- layer 3: per-edge 64-dot (lanes 0..15), write output ----
    if (lane < 16) {
        float s = b3[0];
        const float* hp = sH2 + lane * 64;
        #pragma unroll
        for (int n = 0; n < 64; ++n) s += hp[n] * w3[n];
        int eo = e0 + lane;
        if (eo < E) out[eo] = s;
    }
}

// ---------------- host launch ----------------
extern "C" void kernel_launch(void* const* d_in, const int* in_sizes, int n_in,
                              void* d_out, int out_size, void* d_ws, size_t ws_size,
                              hipStream_t stream) {
    const float* drug = (const float*)d_in[0];
    const float* dis  = (const float*)d_in[1];
    const int*   src  = (const int*)d_in[2];
    const int*   dst  = (const int*)d_in[3];
    const float* W1   = (const float*)d_in[4];
    const float* b1   = (const float*)d_in[5];
    const float* W2   = (const float*)d_in[6];
    const float* b2   = (const float*)d_in[7];
    const float* W3   = (const float*)d_in[8];
    const float* b3   = (const float*)d_in[9];
    float* out = (float*)d_out;
    const int E = in_sizes[2];

    // workspace layout (all 16B aligned)
    char* ws = (char*)d_ws;
    size_t offDis = (size_t)in_sizes[0] * 2;
    size_t offW1  = offDis + (size_t)in_sizes[1] * 2;
    size_t offW2  = offW1 + (size_t)512 * 128 * 2;
    unsigned short* drug_bf = (unsigned short*)(ws);
    unsigned short* dis_bf  = (unsigned short*)(ws + offDis);
    unsigned short* w1t     = (unsigned short*)(ws + offW1);
    unsigned short* w2t     = (unsigned short*)(ws + offW2);

    // prep: feature tables f32 -> bf16
    {
        int n4 = in_sizes[0] / 4;
        cvt_bf16_kernel<<<(n4 + 255) / 256, 256, 0, stream>>>(drug, drug_bf, n4);
        n4 = in_sizes[1] / 4;
        cvt_bf16_kernel<<<(n4 + 255) / 256, 256, 0, stream>>>(dis, dis_bf, n4);
    }
    // prep: weight transposes to bf16
    transpose_bf16_kernel<<<(512 * 128 + 255) / 256, 256, 0, stream>>>(W1, w1t, 512, 128);
    transpose_bf16_kernel<<<(128 * 64 + 255) / 256, 256, 0, stream>>>(W2, w2t, 128, 64);

    // main fused kernel
    (void)hipFuncSetAttribute(reinterpret_cast<const void*>(mlp_edge_kernel),
                              hipFuncAttributeMaxDynamicSharedMemorySize, SM_TOT);
    int blocks = (E + 127) / 128;
    mlp_edge_kernel<<<blocks, 256, SM_TOT, stream>>>(
        drug_bf, dis_bf, src, dst, w1t, w2t, b1, b2, W3, b3, out, E);
}
